// MultiHeadAttention_24120536334625
// MI455X (gfx1250) — matrix operand order repaired and checked by reading, not compiled, not run on hardware
//
#include <hip/hip_runtime.h>

// ---------------------------------------------------------------------------
// MHA forward for B=2, T=2048, E=1024, H=16, hd=64 on gfx1250 (MI455X).
// bf16 WMMA (v_wmma_f32_16x16x32_bf16) with fp32 accumulation everywhere;
// flash-attention (online softmax) so the 536MB score matrix never hits HBM.
// GEMMs are register-blocked 64x64 per wave: 16 WMMAs per 16 b128 loads.
// Flash K/V chunks are staged to LDS with async global->LDS loads (ASYNCcnt),
// double-buffered and shared by all 4 waves of the block.
// ---------------------------------------------------------------------------

typedef __attribute__((ext_vector_type(16))) __bf16 v16bf;
typedef __attribute__((ext_vector_type(8)))  float  v8f;

#define T_DIM  2048
#define E_DIM  1024
#define D_DIM  1024
#define H_NUM  16
#define HD_DIM 64
#define BT_ROWS 4096   // B*T

static __device__ __forceinline__ unsigned short f2bf(float f) {
  unsigned int u = __float_as_uint(f);
  u += 0x7FFFu + ((u >> 16) & 1u);          // round-to-nearest-even
  return (unsigned short)(u >> 16);
}

static __device__ __forceinline__ v8f zero_v8f() {
  v8f z;
#pragma unroll
  for (int i = 0; i < 8; ++i) z[i] = 0.0f;
  return z;
}

static __device__ __forceinline__ v16bf make_frag(uint4 a, uint4 b) {
  union { uint4 u[2]; v16bf v; } U;
  U.u[0] = a; U.u[1] = b;
  return U.v;
}

// A fragment (16x32 bf16, M x K), source row-major with leading dim `ld`.
// Lane l: row = l%16; elems 0..7 = K kb..kb+7, elems 8..15 = K kb+16..kb+23,
// kb = (l<16 ? 0 : 8).   (ISA 7.12.2, 16-bit A 16x32 layout)
static __device__ __forceinline__ v16bf load_a_frag(const unsigned short* base,
                                                    int ld, int lane) {
  const int row = lane & 15;
  const int kb  = (lane < 16) ? 0 : 8;
  const unsigned short* p = base + row * ld + kb;
  uint4 a = *(const uint4*)(p);
  uint4 b = *(const uint4*)(p + 16);
  return make_frag(a, b);
}

// B fragment (32x16 bf16, K x N) loaded from B^T stored row-major (N x K, ld).
// Lane l: col = l%16; elems 0..7 = K kb..kb+7, elems 8..15 = K kb+16..kb+23, kb = (l<16 ? 0 : 8).
static __device__ __forceinline__ v16bf load_bt_frag(const unsigned short* bt,
                                                     int ld, int lane) {
  const int col = lane & 15;
  const int kb  = (lane < 16) ? 0 : 8;
  const unsigned short* p = bt + col * ld + kb;
  uint4 a = *(const uint4*)(p);
  uint4 b = *(const uint4*)(p + 16);
  return make_frag(a, b);
}

static __device__ __forceinline__ v8f wmma_bf16(v16bf a, v16bf b, v8f c) {
  return __builtin_amdgcn_wmma_f32_16x16x32_bf16(
      /*neg_a=*/false, a, /*neg_b=*/false, b,
      /*c_mod=*/(short)0, c, /*reuse_a=*/false, /*reuse_b=*/false);
}

// Stage rows [wave*16, wave*16+16) of a 64-row x 64-elem bf16 tile from global
// to LDS via async global->LDS loads (ASYNCcnt). Each issue moves 4 rows:
// lane -> row (lane>>3), 16-byte column chunk (lane&7).
static __device__ __forceinline__ void async_stage_quarter(
    const unsigned short* gbase, int grsElems,   // global tile base, row stride
    unsigned short* lbase, int wave, int lane) { // LDS tile base (row stride 64)
  const int rsub = lane >> 3;         // 0..3
  const int cb   = (lane & 7) * 8;    // element offset (8 elems = 16B)
#pragma unroll
  for (int q = 0; q < 4; ++q) {
    const int row = wave * 16 + q * 4 + rsub;
    const unsigned short* g = gbase + (size_t)row * grsElems + cb;
    unsigned int lofs = (unsigned int)(uintptr_t)(lbase + row * 64 + cb);
    asm volatile("global_load_async_to_lds_b128 %0, %1, off"
                 :: "v"(lofs), "v"(g) : "memory");
  }
}

// ------------------------------ conversions --------------------------------

__global__ void cvt_bf16_kernel(const float* __restrict__ src,
                                unsigned short* __restrict__ dst, int n) {
  int i = blockIdx.x * blockDim.x + threadIdx.x;
  if (i < n) dst[i] = f2bf(src[i]);
}

// dst[c*rows + r] = bf16(src[r*cols + c])  (store W^T so B-frags are contiguous)
__global__ void transpose_cvt_kernel(const float* __restrict__ src,
                                     unsigned short* __restrict__ dst,
                                     int rows, int cols) {
  int i = blockIdx.x * blockDim.x + threadIdx.x;
  if (i >= rows * cols) return;
  int r = i / cols, c = i - r * cols;
  dst[c * rows + r] = f2bf(src[i]);
}

// ------------------------------ QKV GEMM -----------------------------------
// qkv[4096,3072] = xb[4096,1024] @ W_attn + b_attn, scattered into
//   Q: [B,H,T,64] row-major,  K: [B,H,T,64] row-major,  V^T: [B,H,64,T].
// One wave computes a 64x64 tile (4x4 fragments): 16 WMMAs / 16 loads per k.
__global__ void __launch_bounds__(128)
qkv_gemm_kernel(const unsigned short* __restrict__ xb,
                const unsigned short* __restrict__ wat,   // [3072,1024] = W_attn^T
                const float* __restrict__ b_attn,
                unsigned short* __restrict__ Qb,
                unsigned short* __restrict__ Kb,
                unsigned short* __restrict__ Vt) {
  const int lane = threadIdx.x & 31;
  const int wave = threadIdx.x >> 5;
  const int m0 = blockIdx.y * 64;                      // 64 rows / wave
  const int n0 = blockIdx.x * 256 + wave * 64;         // 64 cols / wave

  v8f acc[4][4];
#pragma unroll
  for (int i = 0; i < 4; ++i)
#pragma unroll
    for (int j = 0; j < 4; ++j) acc[i][j] = zero_v8f();

  for (int k = 0; k < E_DIM; k += 32) {
    v16bf a[4], b[4];
#pragma unroll
    for (int i = 0; i < 4; ++i)
      a[i] = load_a_frag(xb + (m0 + 16 * i) * E_DIM + k, E_DIM, lane);
#pragma unroll
    for (int j = 0; j < 4; ++j)
      b[j] = load_bt_frag(wat + (n0 + 16 * j) * E_DIM + k, E_DIM, lane);
#pragma unroll
    for (int j = 0; j < 4; ++j)
#pragma unroll
      for (int i = 0; i < 4; ++i)
        acc[i][j] = wmma_bf16(a[i], b[j], acc[i][j]);
  }

  const int col = lane & 15;
  const int hiOff = (lane < 16) ? 0 : 8;
#pragma unroll
  for (int j = 0; j < 4; ++j) {
    const int n = n0 + 16 * j + col;
    const float bv = b_attn[n];
    const int part = n >> 10;            // 0:q 1:k 2:v (uniform per fragment)
    const int h = (n & 1023) >> 6;
    const int d = n & 63;
#pragma unroll
    for (int i = 0; i < 4; ++i) {
      const int rbase = m0 + 16 * i + hiOff;
#pragma unroll
      for (int v = 0; v < 8; ++v) {
        const int r  = rbase + v;        // global row = b*T + t
        const int bb = r >> 11;
        const int t  = r & (T_DIM - 1);
        const unsigned short o = f2bf(acc[i][j][v] + bv);
        if (part == 0)
          Qb[(((bb * H_NUM) + h) * T_DIM + t) * HD_DIM + d] = o;
        else if (part == 1)
          Kb[(((bb * H_NUM) + h) * T_DIM + t) * HD_DIM + d] = o;
        else
          Vt[(((bb * H_NUM) + h) * HD_DIM + d) * T_DIM + t] = o;
      }
    }
  }
}

// --------------------------- flash attention -------------------------------
// grid = (T/64, B*H), block = 128 (4 waves). The block cooperatively stages
// 64-key K/V chunks into LDS with async global->LDS loads (double-buffered);
// every wave owns 16 query rows. All waves run to the block's causal limit so
// the block barrier is uniform; early waves see fully-masked (inert) tiles.
__global__ void __launch_bounds__(128)
flash_attn_kernel(const unsigned short* __restrict__ Qb,
                  const unsigned short* __restrict__ Kb,
                  const unsigned short* __restrict__ Vt,
                  unsigned short* __restrict__ Yb) {
  __shared__ __align__(16) unsigned short kbuf[2][64][64];   // 2 x 8KB
  __shared__ __align__(16) unsigned short vbuf[2][64][64];   // 2 x 8KB
  __shared__ __align__(16) unsigned short plds[4][16][64];   // 8KB

  const int lane = threadIdx.x & 31;
  const int wave = threadIdx.x >> 5;
  const int bh = blockIdx.y;                       // b*H + h
  const int q0 = blockIdx.x * 64;                  // block's first query row
  const int qw = q0 + wave * 16;                   // wave's first query row

  const unsigned short* Q = Qb + (bh * T_DIM + qw) * HD_DIM;
  const unsigned short* K = Kb + bh * T_DIM * HD_DIM;
  const unsigned short* V = Vt + bh * HD_DIM * T_DIM;

  const v16bf qa0 = load_a_frag(Q + 0,  HD_DIM, lane);   // hd 0..31
  const v16bf qa1 = load_a_frag(Q + 32, HD_DIM, lane);   // hd 32..63

  v8f o[4];
#pragma unroll
  for (int j = 0; j < 4; ++j) o[j] = zero_v8f();

  float mrow[8], lrow[8];
#pragma unroll
  for (int v = 0; v < 8; ++v) { mrow[v] = -1e30f; lrow[v] = 0.0f; }

  const int hi  = (lane < 16) ? 0 : 8;
  const int col = lane & 15;
  const int kend = q0 + 64;                        // uniform block causal limit

  // prologue: stage chunk 0 into buffer 0
  async_stage_quarter(K, HD_DIM, &kbuf[0][0][0], wave, lane);
  async_stage_quarter(V, T_DIM,  &vbuf[0][0][0], wave, lane);

  int cur = 0;
  for (int kt = 0; kt < kend; kt += 64, cur ^= 1) {
    // wait for own quarter of chunk kt, then barrier => whole chunk ready
    asm volatile("s_wait_asynccnt 0x0" ::: "memory");
    __syncthreads();
    if (kt + 64 < kend) {   // stage next chunk into the other buffer
      async_stage_quarter(K + (kt + 64) * HD_DIM, HD_DIM,
                          &kbuf[cur ^ 1][0][0], wave, lane);
      async_stage_quarter(V + (kt + 64), T_DIM,
                          &vbuf[cur ^ 1][0][0], wave, lane);
    }
    const unsigned short* Kl = &kbuf[cur][0][0];
    const unsigned short* Vl = &vbuf[cur][0][0];

    // ---- S = Q K^T for 16 rows x 64 keys (4 tiles x 2 K-steps) ----
    v8f s[4];
#pragma unroll
    for (int tt = 0; tt < 4; ++tt) {
      s[tt] = zero_v8f();
      v16bf b0 = load_bt_frag(Kl + (16 * tt) * 64 + 0,  64, lane);
      v16bf b1 = load_bt_frag(Kl + (16 * tt) * 64 + 32, 64, lane);
      s[tt] = wmma_bf16(qa0, b0, s[tt]);
      s[tt] = wmma_bf16(qa1, b1, s[tt]);
    }

    // ---- scale + causal mask + online softmax (per-row, half-wave) ----
    float corr[8];
#pragma unroll
    for (int v = 0; v < 8; ++v) {
      const int rowg = qw + v + hi;
      float a[4];
#pragma unroll
      for (int tt = 0; tt < 4; ++tt) {
        a[tt] = s[tt][v] * 0.125f;                 // 1/sqrt(64)
        if (kt + 16 * tt + col > rowg) a[tt] = -1e30f;
      }
      float m = fmaxf(fmaxf(a[0], a[1]), fmaxf(a[2], a[3]));
#pragma unroll
      for (int off = 1; off < 16; off <<= 1) m = fmaxf(m, __shfl_xor(m, off, 32));
      const float mn = fmaxf(mrow[v], m);
      corr[v] = __expf(mrow[v] - mn);
      float rs = 0.0f;
#pragma unroll
      for (int tt = 0; tt < 4; ++tt) {
        const float p = __expf(a[tt] - mn);
        s[tt][v] = p;
        rs += p;
      }
#pragma unroll
      for (int off = 1; off < 16; off <<= 1) rs += __shfl_xor(rs, off, 32);
      lrow[v] = lrow[v] * corr[v] + rs;
      mrow[v] = mn;
    }
#pragma unroll
    for (int j = 0; j < 4; ++j)
#pragma unroll
      for (int v = 0; v < 8; ++v) o[j][v] *= corr[v];

    // ---- C-fragment -> A-fragment relayout of P through LDS ----
#pragma unroll
    for (int tt = 0; tt < 4; ++tt)
#pragma unroll
      for (int v = 0; v < 8; ++v)
        plds[wave][v + hi][16 * tt + col] = f2bf(s[tt][v]);
    asm volatile("s_wait_dscnt 0" ::: "memory");    // wave-local LDS RAW fence
    const v16bf pa0 = load_a_frag(&plds[wave][0][0],      64, lane);
    const v16bf pa1 = load_a_frag(&plds[wave][0][0] + 32, 64, lane);
    asm volatile("s_wait_dscnt 0" ::: "memory");    // loads done before rewrite

    // ---- O += P @ V (64 keys x 64 hd) ----
#pragma unroll
    for (int j = 0; j < 4; ++j) {
      v16bf vb0 = load_bt_frag(Vl + (16 * j) * 64 + 0,  64, lane);
      v16bf vb1 = load_bt_frag(Vl + (16 * j) * 64 + 32, 64, lane);
      o[j] = wmma_bf16(pa0, vb0, o[j]);
      o[j] = wmma_bf16(pa1, vb1, o[j]);
    }
  }

  // ---- normalize and store y as bf16 [B,T,D] ----
  float linv[8];
#pragma unroll
  for (int v = 0; v < 8; ++v) linv[v] = 1.0f / lrow[v];
  const int bb = bh >> 4;
  const int h  = bh & 15;
#pragma unroll
  for (int j = 0; j < 4; ++j)
#pragma unroll
    for (int v = 0; v < 8; ++v) {
      const int t = qw + v + hi;
      Yb[(bb * T_DIM + t) * D_DIM + h * HD_DIM + 16 * j + col] =
          f2bf(o[j][v] * linv[v]);
    }
}

// ---------------------------- projection GEMM ------------------------------
// out[4096,1024] f32 = Yb[4096,1024]bf16 @ W_proj + b_proj; 64x64 per wave.
__global__ void __launch_bounds__(128)
proj_gemm_kernel(const unsigned short* __restrict__ yb,
                 const unsigned short* __restrict__ wpt,  // [1024,1024] = W_proj^T
                 const float* __restrict__ b_proj,
                 float* __restrict__ out) {
  const int lane = threadIdx.x & 31;
  const int wave = threadIdx.x >> 5;
  const int m0 = blockIdx.y * 64;
  const int n0 = blockIdx.x * 256 + wave * 64;

  v8f acc[4][4];
#pragma unroll
  for (int i = 0; i < 4; ++i)
#pragma unroll
    for (int j = 0; j < 4; ++j) acc[i][j] = zero_v8f();

  for (int k = 0; k < D_DIM; k += 32) {
    v16bf a[4], b[4];
#pragma unroll
    for (int i = 0; i < 4; ++i)
      a[i] = load_a_frag(yb + (m0 + 16 * i) * D_DIM + k, D_DIM, lane);
#pragma unroll
    for (int j = 0; j < 4; ++j)
      b[j] = load_bt_frag(wpt + (n0 + 16 * j) * D_DIM + k, D_DIM, lane);
#pragma unroll
    for (int j = 0; j < 4; ++j)
#pragma unroll
      for (int i = 0; i < 4; ++i)
        acc[i][j] = wmma_bf16(a[i], b[j], acc[i][j]);
  }

  const int col = lane & 15;
  const int hiOff = (lane < 16) ? 0 : 8;
#pragma unroll
  for (int j = 0; j < 4; ++j) {
    const int n = n0 + 16 * j + col;
    const float bv = b_proj[n];
#pragma unroll
    for (int i = 0; i < 4; ++i) {
      const int rbase = m0 + 16 * i + hiOff;
#pragma unroll
      for (int v = 0; v < 8; ++v)
        out[(rbase + v) * E_DIM + n] = acc[i][j][v] + bv;
    }
  }
}

// ------------------------------- launcher ----------------------------------
extern "C" void kernel_launch(void* const* d_in, const int* in_sizes, int n_in,
                              void* d_out, int out_size, void* d_ws, size_t ws_size,
                              hipStream_t stream) {
  const float* x      = (const float*)d_in[0];   // [B,T,E]
  const float* W_attn = (const float*)d_in[1];   // [E,3D]
  const float* b_attn = (const float*)d_in[2];   // [3D]
  const float* W_proj = (const float*)d_in[3];   // [D,E]
  const float* b_proj = (const float*)d_in[4];   // [E]
  float* out = (float*)d_out;                    // [B,T,E] f32

  char* ws = (char*)d_ws;
  size_t off = 0;
  auto carve = [&](size_t bytes) {
    char* p = ws + off;
    off = (off + bytes + 255) & ~(size_t)255;
    return p;
  };
  unsigned short* xb  = (unsigned short*)carve((size_t)BT_ROWS * E_DIM * 2);    // 8 MB
  unsigned short* wat = (unsigned short*)carve((size_t)3 * D_DIM * E_DIM * 2);  // 6 MB
  unsigned short* wpt = (unsigned short*)carve((size_t)D_DIM * E_DIM * 2);      // 2 MB
  unsigned short* Qb  = (unsigned short*)carve((size_t)BT_ROWS * D_DIM * 2);    // 8 MB
  unsigned short* Kb  = (unsigned short*)carve((size_t)BT_ROWS * D_DIM * 2);    // 8 MB
  unsigned short* Vt  = (unsigned short*)carve((size_t)BT_ROWS * D_DIM * 2);    // 8 MB
  unsigned short* Yb  = (unsigned short*)carve((size_t)BT_ROWS * D_DIM * 2);    // 8 MB

  {
    int n = BT_ROWS * E_DIM;
    cvt_bf16_kernel<<<(n + 255) / 256, 256, 0, stream>>>(x, xb, n);
  }
  {
    int n = E_DIM * 3 * D_DIM;
    transpose_cvt_kernel<<<(n + 255) / 256, 256, 0, stream>>>(W_attn, wat, E_DIM, 3 * D_DIM);
  }
  {
    int n = D_DIM * E_DIM;
    transpose_cvt_kernel<<<(n + 255) / 256, 256, 0, stream>>>(W_proj, wpt, D_DIM, E_DIM);
  }

  qkv_gemm_kernel<<<dim3(3 * D_DIM / 256, BT_ROWS / 64), 128, 0, stream>>>(
      xb, wat, b_attn, Qb, Kb, Vt);

  flash_attn_kernel<<<dim3(T_DIM / 64, 2 * H_NUM), 128, 0, stream>>>(Qb, Kb, Vt, Yb);

  proj_gemm_kernel<<<dim3(E_DIM / 256, BT_ROWS / 64), 128, 0, stream>>>(
      Yb, wpt, b_proj, out);
}
